// HybridAoAGNN_33114197852578
// MI455X (gfx1250) — matrix operand order repaired and checked by reading, not compiled, not run on hardware
//
#include <hip/hip_runtime.h>
#include <hip/hip_bf16.h>

// ---------------------------------------------------------------------------
// Problem constants (match reference)
// ---------------------------------------------------------------------------
#define NN   50000
#define EE   300000
#define INF_ 64
#define HH   256
#define LL   4
#define NBB  181
#define GG   64
#define NEG_SLOPE 0.2f
#define LN_EPS 1e-5f

typedef __attribute__((ext_vector_type(16))) __bf16 v16bf;
typedef __attribute__((ext_vector_type(8)))  float  v8f;

// ---------------------------------------------------------------------------
// Helpers
// ---------------------------------------------------------------------------
__device__ __forceinline__ unsigned bf16rne(float f) {
    union { unsigned u; float f; } in; in.f = f;
    unsigned u = in.u;
    return (u + 0x7FFFu + ((u >> 16) & 1u)) >> 16;     // round-to-nearest-even
}
// pack two floats into one dword of bf16 (lo in [15:0], hi in [31:16])
__device__ __forceinline__ unsigned pack2bf(float lo, float hi) {
    return bf16rne(lo) | (bf16rne(hi) << 16);
}

// order-preserving float<->uint for atomic max
__device__ __forceinline__ unsigned f2ord(float f) {
    unsigned u = __float_as_uint(f);
    return (u & 0x80000000u) ? ~u : (u | 0x80000000u);
}
__device__ __forceinline__ float ord2f(unsigned e) {
    unsigned u = (e & 0x80000000u) ? (e ^ 0x80000000u) : ~e;
    return __uint_as_float(u);
}

__device__ __forceinline__ float wave_sum(float s) {
    #pragma unroll
    for (int off = 16; off; off >>= 1) s += __shfl_xor(s, off, 32);
    return s;
}

// ---------------------------------------------------------------------------
// WMMA GEMM:  C[M,Nout] = A[M,K] @ W[Nout,K]^T + bias   (optional ReLU)
// Block = 256 thr (8 waves).  Block tile 64(M) x 128(N); wave tile 16 x 64.
// Requirements (guaranteed by launcher): K % 32 == 0, Nout % 128 == 0,
// A and W 16-byte aligned rows (K multiple of 4 floats).
// f32 inputs staged to LDS as bf16 via b128 loads + packed b32 stores;
// accumulate f32 with v_wmma_f32_16x16x32_bf16.
// ---------------------------------------------------------------------------
#define TM 64
#define TN 128
#define KC 32
#define LDP (KC + 2)      // bf16 elements per LDS row (padded: 17 dwords)

__global__ __launch_bounds__(256)
void gemm_bf16_wmma(const float* __restrict__ A, const float* __restrict__ W,
                    const float* __restrict__ bias, float* __restrict__ C,
                    int M, int K, int Nout, int act)
{
    __shared__ __bf16 As[TM][LDP];
    __shared__ __bf16 Bs[TN][LDP];

    const int tid  = threadIdx.x;
    const int bm   = blockIdx.x * TM;
    const int bn   = blockIdx.y * TN;
    const int wave = tid >> 5;
    const int lane = tid & 31;
    const int wm   = wave & 3;       // wave row group (4)
    const int wn   = wave >> 2;      // wave col group (2)
    const int half = lane >> 4;      // lane half (0: lanes0-15, 1: lanes16-31)
    const int l16  = lane & 15;

    // precompute staging coordinates (float4 granularity: KC/4 = 8 groups/row)
    // A: 64 rows x 8 groups = 512 units, 2 per thread
    const int ar0 = tid >> 3,            ac0 = (tid & 7) * 4;        // unit tid
    const int ar1 = (tid + 256) >> 3,    ac1 = ac0;                  // unit tid+256
    // B: 128 rows x 8 groups = 1024 units, 4 per thread
    const int br_ = tid >> 3;            // rows tid>>3 + {0,32,64,96}
    const int bc_ = (tid & 7) * 4;

    // clamped global rows for A (values for rows >= M are discarded in epilogue)
    const int garow0 = min(bm + ar0, M - 1);
    const int garow1 = min(bm + ar1, M - 1);

    v8f acc[4];
    #pragma unroll
    for (int t = 0; t < 4; ++t)
        #pragma unroll
        for (int j = 0; j < 8; ++j) acc[t][j] = 0.0f;

    for (int k0 = 0; k0 < K; k0 += KC) {
        // ---- prefetch next K-chunk (global_prefetch_b8) ----
        if (k0 + KC < K) {
            __builtin_prefetch(A + (long long)garow0 * K + k0 + KC + ac0, 0, 1);
            __builtin_prefetch(W + (long long)(bn + br_) * K + k0 + KC + bc_, 0, 1);
        }

        // ---- stage A tile (b128 load -> 2x b32 packed bf16 store) ----
        {
            float4 v = *(const float4*)(A + (long long)garow0 * K + k0 + ac0);
            unsigned* p = (unsigned*)&As[ar0][ac0];
            p[0] = pack2bf(v.x, v.y);
            p[1] = pack2bf(v.z, v.w);
            v = *(const float4*)(A + (long long)garow1 * K + k0 + ac1);
            p = (unsigned*)&As[ar1][ac1];
            p[0] = pack2bf(v.x, v.y);
            p[1] = pack2bf(v.z, v.w);
        }
        // ---- stage B tile (W rows; Nout % 128 == 0, no guard) ----
        #pragma unroll
        for (int u = 0; u < 4; ++u) {
            int r = br_ + u * 32;
            float4 v = *(const float4*)(W + (long long)(bn + r) * K + k0 + bc_);
            unsigned* p = (unsigned*)&Bs[r][bc_];
            p[0] = pack2bf(v.x, v.y);
            p[1] = pack2bf(v.z, v.w);
        }
        __syncthreads();

        // ---- A fragment: 16x32 bf16 per ISA layout ----
        v16bf af;
        const int am = wm * 16 + l16;
        #pragma unroll
        for (int v = 0; v < 8; ++v) {
            int kk = ((v < 4) ? 2 * v : 2 * v + 8) + half * 8;
            af[2 * v]     = As[am][kk];
            af[2 * v + 1] = As[am][kk + 1];
        }
        // ---- 4 WMMA tiles along N ----
        #pragma unroll
        for (int t = 0; t < 4; ++t) {
            v16bf bfm;
            const int bcol = wn * 64 + t * 16 + l16;
            #pragma unroll
            for (int v = 0; v < 8; ++v) {
                int kk = ((v < 4) ? 2 * v : 2 * v + 8) + half * 8;
                bfm[2 * v]     = Bs[bcol][kk];
                bfm[2 * v + 1] = Bs[bcol][kk + 1];
            }
            acc[t] = __builtin_amdgcn_wmma_f32_16x16x32_bf16(
                         false, af, false, bfm, (short)0, acc[t], false, false);
        }
        __syncthreads();
    }

    // epilogue: C layout — VGPR v holds row (v + half*8), col = lane&15
    #pragma unroll
    for (int t = 0; t < 4; ++t) {
        int col = bn + wn * 64 + t * 16 + l16;
        float bcol = bias[col];
        #pragma unroll
        for (int v = 0; v < 8; ++v) {
            int row = bm + wm * 16 + v + half * 8;
            if (row < M) {
                float r = acc[t][v] + bcol;
                if (act) r = fmaxf(r, 0.0f);
                C[(long long)row * Nout + col] = r;
            }
        }
    }
}

// ---------------------------------------------------------------------------
// Edge pass 1: e = leaky_relu(xl[src]+xr[dst]) . att ; segment max over dst
// one wave per edge
// ---------------------------------------------------------------------------
__global__ void edge_score_max(const float* __restrict__ xl, const float* __restrict__ xr,
                               const float* __restrict__ att,
                               const long long* __restrict__ ei,
                               int tot, float* __restrict__ ebuf, unsigned* __restrict__ mbuf)
{
    int w = (int)((blockIdx.x * (unsigned)blockDim.x + threadIdx.x) >> 5);
    int lane = threadIdx.x & 31;
    if (w >= tot) return;
    long long s, d;
    if (w < EE) { s = ei[w]; d = ei[EE + w]; } else { s = d = w - EE; }
    const float* pl = xl + s * HH;
    const float* pr = xr + d * HH;
    float sum = 0.0f;
    #pragma unroll
    for (int k = lane; k < HH; k += 32) {
        float h = pl[k] + pr[k];
        h = (h > 0.0f) ? h : h * NEG_SLOPE;
        sum += h * att[k];
    }
    sum = wave_sum(sum);
    if (lane == 0) {
        ebuf[w] = sum;
        atomicMax(mbuf + d, f2ord(sum));
    }
}

// Edge pass 2: a = exp(e - m[dst]); d[dst] += a  (one thread per edge)
__global__ void edge_exp_sum(const long long* __restrict__ ei, int tot,
                             float* __restrict__ ebuf, const unsigned* __restrict__ mbuf,
                             float* __restrict__ dbuf)
{
    int e = blockIdx.x * blockDim.x + threadIdx.x;
    if (e >= tot) return;
    long long d = (e < EE) ? ei[EE + e] : (long long)(e - EE);
    float a = __expf(ebuf[e] - ord2f(mbuf[d]));
    ebuf[e] = a;
    atomicAdd(dbuf + d, a);
}

// Edge pass 3: agg[dst] += (a/d[dst]) * xl[src]  (one wave per edge)
__global__ void edge_scatter(const float* __restrict__ xl,
                             const long long* __restrict__ ei, int tot,
                             const float* __restrict__ ebuf, const float* __restrict__ dbuf,
                             float* __restrict__ agg)
{
    int w = (int)((blockIdx.x * (unsigned)blockDim.x + threadIdx.x) >> 5);
    int lane = threadIdx.x & 31;
    if (w >= tot) return;
    long long s, d;
    if (w < EE) { s = ei[w]; d = ei[EE + w]; } else { s = d = w - EE; }
    float coef = ebuf[w] / dbuf[d];
    const float* ps = xl + s * HH;
    float* pd = agg + d * HH;
    #pragma unroll
    for (int k = lane; k < HH; k += 32)
        atomicAdd(pd + k, coef * ps[k]);
}

// Fused: out = relu(LN(agg + bias_c)*g + b) + x_res ; write to x (in place)
// one block (256 thr) per node
__global__ __launch_bounds__(256)
void ln_relu_residual(const float* __restrict__ agg, const float* __restrict__ bias_c,
                      const float* __restrict__ g, const float* __restrict__ b,
                      float* __restrict__ x)
{
    __shared__ float red[2][8];
    int n = blockIdx.x, t = threadIdx.x;
    float v = agg[(long long)n * HH + t] + bias_c[t];
    float s = wave_sum(v), s2 = wave_sum(v * v);
    int wv = t >> 5, lane = t & 31;
    if (lane == 0) { red[0][wv] = s; red[1][wv] = s2; }
    __syncthreads();
    if (t == 0) {
        float a = 0.f, c = 0.f;
        #pragma unroll
        for (int i = 0; i < 8; ++i) { a += red[0][i]; c += red[1][i]; }
        red[0][0] = a; red[1][0] = c;
    }
    __syncthreads();
    float mean = red[0][0] * (1.0f / HH);
    float var  = red[1][0] * (1.0f / HH) - mean * mean;
    float inv  = rsqrtf(var + LN_EPS);
    float o = (v - mean) * inv * g[t] + b[t];
    o = fmaxf(o, 0.0f);
    x[(long long)n * HH + t] += o;
}

// matvec: out[n] = dot(A[n,:], w) + b0  (one wave per row)
__global__ void matvec_dot(const float* __restrict__ A, const float* __restrict__ w,
                           const float* __restrict__ b0, float* __restrict__ out,
                           int M, int K)
{
    int r = (int)((blockIdx.x * (unsigned)blockDim.x + threadIdx.x) >> 5);
    int lane = threadIdx.x & 31;
    if (r >= M) return;
    float s = 0.0f;
    for (int k = lane; k < K; k += 32) s += A[(long long)r * K + k] * w[k];
    s = wave_sum(s);
    if (lane == 0) out[r] = s + b0[0];
}

// pooling softmax pass 1: per-graph max of gate
__global__ void pool_max(const float* __restrict__ gate, const long long* __restrict__ batch,
                         unsigned* __restrict__ m2)
{
    int n = blockIdx.x * blockDim.x + threadIdx.x;
    if (n >= NN) return;
    atomicMax(m2 + batch[n], f2ord(gate[n]));
}

// pooling softmax pass 2: a = exp(gate - m); d2 += a
__global__ void pool_exp_sum(float* __restrict__ gate, const long long* __restrict__ batch,
                             const unsigned* __restrict__ m2, float* __restrict__ d2)
{
    int n = blockIdx.x * blockDim.x + threadIdx.x;
    if (n >= NN) return;
    long long gidx = batch[n];
    float a = __expf(gate[n] - ord2f(m2[gidx]));
    gate[n] = a;
    atomicAdd(d2 + gidx, a);
}

// pooling pass 3: pooled[g] += (a/d2[g]) * x[n]  (one wave per node)
__global__ void pool_scatter(const float* __restrict__ x, const float* __restrict__ gate,
                             const long long* __restrict__ batch, const float* __restrict__ d2,
                             float* __restrict__ pooled)
{
    int n = (int)((blockIdx.x * (unsigned)blockDim.x + threadIdx.x) >> 5);
    int lane = threadIdx.x & 31;
    if (n >= NN) return;
    long long gidx = batch[n];
    float coef = gate[n] / d2[gidx];
    const float* ps = x + (long long)n * HH;
    float* pd = pooled + gidx * HH;
    #pragma unroll
    for (int k = lane; k < HH; k += 32)
        atomicAdd(pd + k, coef * ps[k]);
}

// classifier head: out[g,c] = z[g,:] . Wc[c,:] + bc[c]  (thread per output)
__global__ void cls_head(const float* __restrict__ z, const float* __restrict__ Wc,
                         const float* __restrict__ bc, float* __restrict__ out)
{
    int i = blockIdx.x * blockDim.x + threadIdx.x;
    if (i >= GG * NBB) return;
    int g = i / NBB, c = i % NBB;
    float s = bc[c];
    const float* pz = z + g * HH;
    const float* pw = Wc + c * HH;
    #pragma unroll 4
    for (int k = 0; k < HH; ++k) s += pz[k] * pw[k];
    out[i] = s;
}

// residual head: out[g] = tanh(r1[g,:] . Wr2 + br2)  (one wave per graph)
__global__ void resid_head(const float* __restrict__ r1, const float* __restrict__ Wr2,
                           const float* __restrict__ br2, float* __restrict__ out)
{
    int g = (int)((blockIdx.x * (unsigned)blockDim.x + threadIdx.x) >> 5);
    int lane = threadIdx.x & 31;
    if (g >= GG) return;
    float s = 0.0f;
    for (int k = lane; k < 128; k += 32) s += r1[g * 128 + k] * Wr2[k];
    s = wave_sum(s);
    if (lane == 0) out[g] = tanhf(s + br2[0]);
}

// ---------------------------------------------------------------------------
// Host orchestration
// ---------------------------------------------------------------------------
extern "C" void kernel_launch(void* const* d_in, const int* in_sizes, int n_in,
                              void* d_out, int out_size, void* d_ws, size_t ws_size,
                              hipStream_t stream)
{
    const float*     x_in  = (const float*)d_in[0];
    const long long* ei    = (const long long*)d_in[1];
    const long long* batch = (const long long*)d_in[2];
    const float* W_in = (const float*)d_in[3];   const float* b_in = (const float*)d_in[4];
    const float* Wl   = (const float*)d_in[5];   const float* bl   = (const float*)d_in[6];
    const float* Wr   = (const float*)d_in[7];   const float* br   = (const float*)d_in[8];
    const float* att  = (const float*)d_in[9];   const float* bias_c = (const float*)d_in[10];
    const float* ln_g = (const float*)d_in[11];  const float* ln_b = (const float*)d_in[12];
    const float* Wg1  = (const float*)d_in[13];  const float* bg1  = (const float*)d_in[14];
    const float* Wg2  = (const float*)d_in[15];  const float* bg2  = (const float*)d_in[16];
    const float* Ws   = (const float*)d_in[17];  const float* bs   = (const float*)d_in[18];
    const float* Wc   = (const float*)d_in[19];  const float* bc   = (const float*)d_in[20];
    const float* Wr1  = (const float*)d_in[21];  const float* br1  = (const float*)d_in[22];
    const float* Wr2  = (const float*)d_in[23];  const float* br2  = (const float*)d_in[24];

    float* out = (float*)d_out;

    // ---- workspace carve-up (floats) ----
    const long long NH = (long long)NN * HH;
    float* ws = (float*)d_ws;
    float*    x    = ws;                 // [N,H]
    float*    xl   = x   + NH;           // [N,H]  (reused as g1)
    float*    xr   = xl  + NH;           // [N,H]
    float*    agg  = xr  + NH;           // [N,H]
    float*    ebuf = agg + NH;           // [E+N]
    unsigned* mbuf = (unsigned*)(ebuf + (EE + NN));  // [N]
    float*    dbuf = (float*)(mbuf + NN);            // [N]
    float*    gate = dbuf + NN;                      // [N]
    unsigned* m2   = (unsigned*)(gate + NN);         // [G]
    float*    d2   = (float*)(m2 + GG);              // [G]
    float*    pooled = d2 + GG;                      // [G,H]
    float*    z    = pooled + (long long)GG * HH;    // [G,H]
    float*    r1   = z + (long long)GG * HH;         // [G,128]

    const int tot = EE + NN;            // edges incl. self-loops
    dim3 blk(256);

    // 1) input projection: x = x_in @ W_in^T + b_in   [N,64]->[N,256]
    {
        dim3 grid((NN + TM - 1) / TM, HH / TN);
        gemm_bf16_wmma<<<grid, blk, 0, stream>>>(x_in, W_in, b_in, x, NN, INF_, HH, 0);
    }

    const int edgeWaveBlocks = (tot + 7) / 8;    // 8 waves per 256-thr block
    const int nodeWaveBlocks = (NN + 7) / 8;

    // 2) GATv2 layers
    for (int l = 0; l < LL; ++l) {
        dim3 grid((NN + TM - 1) / TM, HH / TN);
        gemm_bf16_wmma<<<grid, blk, 0, stream>>>(x, Wl + (long long)l * HH * HH,
                                                 bl + l * HH, xl, NN, HH, HH, 0);
        gemm_bf16_wmma<<<grid, blk, 0, stream>>>(x, Wr + (long long)l * HH * HH,
                                                 br + l * HH, xr, NN, HH, HH, 0);

        hipMemsetAsync(mbuf, 0, NN * sizeof(unsigned), stream);   // ordered-min
        hipMemsetAsync(dbuf, 0, NN * sizeof(float), stream);
        hipMemsetAsync(agg,  0, NH * sizeof(float), stream);

        edge_score_max<<<edgeWaveBlocks, blk, 0, stream>>>(xl, xr, att + l * HH,
                                                           ei, tot, ebuf, mbuf);
        edge_exp_sum<<<(tot + 255) / 256, blk, 0, stream>>>(ei, tot, ebuf, mbuf, dbuf);
        edge_scatter<<<edgeWaveBlocks, blk, 0, stream>>>(xl, ei, tot, ebuf, dbuf, agg);

        ln_relu_residual<<<NN, blk, 0, stream>>>(agg, bias_c + l * HH,
                                                 ln_g + l * HH, ln_b + l * HH, x);
    }

    // 3) gate MLP:  g1 = relu(x@Wg1^T+bg1);  gate = g1@Wg2^T+bg2
    {
        dim3 grid((NN + TM - 1) / TM, HH / TN);
        gemm_bf16_wmma<<<grid, blk, 0, stream>>>(x, Wg1, bg1, xl /*=g1*/, NN, HH, HH, 1);
        matvec_dot<<<nodeWaveBlocks, blk, 0, stream>>>(xl, Wg2, bg2, gate, NN, HH);
    }

    // 4) attention pooling over batch
    hipMemsetAsync(m2, 0, GG * sizeof(unsigned), stream);
    hipMemsetAsync(d2, 0, GG * sizeof(float), stream);
    hipMemsetAsync(pooled, 0, (long long)GG * HH * sizeof(float), stream);
    pool_max<<<(NN + 255) / 256, blk, 0, stream>>>(gate, batch, m2);
    pool_exp_sum<<<(NN + 255) / 256, blk, 0, stream>>>(gate, batch, m2, d2);
    pool_scatter<<<nodeWaveBlocks, blk, 0, stream>>>(x, gate, batch, d2, pooled);

    // 5) heads:  z = relu(pooled@Ws^T+bs)
    {
        dim3 grid((GG + TM - 1) / TM, HH / TN);
        gemm_bf16_wmma<<<grid, blk, 0, stream>>>(pooled, Ws, bs, z, GG, HH, HH, 1);
    }
    cls_head<<<(GG * NBB + 255) / 256, blk, 0, stream>>>(z, Wc, bc, out);

    {
        dim3 grid((GG + TM - 1) / TM, 128 / TN);   // Nout=128 -> 1 col block
        gemm_bf16_wmma<<<grid, blk, 0, stream>>>(z, Wr1, br1, r1, GG, HH, 128, 1);
    }
    resid_head<<<(GG + 7) / 8, blk, 0, stream>>>(r1, Wr2, br2, out + GG * NBB);
}
